// HyperbolicGNN_5823975653996
// MI455X (gfx1250) — matrix operand order repaired and checked by reading, not compiled, not run on hardware
//
#include <hip/hip_runtime.h>

// ---------------------------------------------------------------------------
// Types / helpers
// ---------------------------------------------------------------------------
typedef __attribute__((ext_vector_type(16))) __bf16 v16bf;
typedef __attribute__((ext_vector_type(8)))  float  v8f;
typedef unsigned uv4 __attribute__((ext_vector_type(4)));

union Frag { uv4 q[2]; v16bf v; };

__device__ __forceinline__ float bf2f(unsigned short h) {
    unsigned u = ((unsigned)h) << 16;
    float f;
    __builtin_memcpy(&f, &u, 4);
    return f;
}
__device__ __forceinline__ unsigned short f2bf(float f) {
    unsigned u;
    __builtin_memcpy(&u, &f, 4);
    u += 0x7FFFu + ((u >> 16) & 1u);   // round-to-nearest-even
    return (unsigned short)(u >> 16);
}
// pack two f32 -> one dword of 2x bf16 (hardware packed cvt when available)
__device__ __forceinline__ unsigned pack2bf(float lo, float hi) {
#if __has_builtin(__builtin_amdgcn_cvt_pk_bf16_f32)
    typedef __bf16 bf2 __attribute__((ext_vector_type(2)));
    bf2 r = __builtin_amdgcn_cvt_pk_bf16_f32(lo, hi);
    unsigned u;
    __builtin_memcpy(&u, &r, 4);
    return u;
#else
    return (unsigned)f2bf(lo) | ((unsigned)f2bf(hi) << 16);
#endif
}

// LDS byte offset of a __shared__ pointer (for async-to-LDS VDST operand)
__device__ __forceinline__ unsigned ldsoff(const void* p) {
    return (unsigned)(unsigned long long)(const __attribute__((address_space(3))) void*)p;
}
// CDNA5 async gather: global -> LDS, tracked by ASYNCcnt (bypasses VGPRs)
__device__ __forceinline__ void async_ld_b128(unsigned lds_byte, const void* gaddr) {
    asm volatile("global_load_async_to_lds_b128 %0, %1, off"
                 :: "v"(lds_byte), "v"(gaddr) : "memory");
}
__device__ __forceinline__ void wait_async0() {
    asm volatile("s_wait_asynccnt 0" ::: "memory");
}

// ---------------------------------------------------------------------------
// Problem constants
// ---------------------------------------------------------------------------
#define N_NODES  30000
#define N_EDGES  480000
#define N_GRAPHS 128
#define N_CENT   100
#define D_FIN    128

// workspace layout (bytes)
static const size_t OFF_XBA   = 0;                         // bf16 [N,128] ping
static const size_t OFF_XBB   = 7680000;                   // bf16 [N,128] pong
static const size_t OFF_ACC   = 15360000;                  // f32  [N,128]
static const size_t OFF_DEG   = 30720000;                  // f32  [N] deg -> deg_inv
static const size_t OFF_WF    = 30840064;                  // weight fragments, 128KB/layer
static const size_t OFF_STATS = OFF_WF + 6 * 131072;       // 32 floats
static const size_t OFF_POOL  = OFF_STATS + 256;           // f32 [128,100]
static const size_t OFF_CNT   = OFF_POOL + 128 * 100 * 4;  // f32 [128]

// ---------------------------------------------------------------------------
// Small utility kernels
// ---------------------------------------------------------------------------
__global__ void k_zero_f32(float* __restrict__ p, int n) {
    int i = blockIdx.x * blockDim.x + threadIdx.x;
    if (i < n) p[i] = 0.0f;
}

__global__ void k_deg(const int* __restrict__ dst, float* __restrict__ deg, int E) {
    int e = blockIdx.x * blockDim.x + threadIdx.x;
    if (e < E) atomicAdd(&deg[dst[e]], 1.0f);
}

// one block, 256 threads: batch-norm statistics over [N,5]
__global__ void k_bn_stats(const float* __restrict__ x, const float* __restrict__ g,
                           const float* __restrict__ b, float* __restrict__ stats, int N) {
    __shared__ float red[256];
    float s[5] = {0, 0, 0, 0, 0}, q[5] = {0, 0, 0, 0, 0};
    for (int n = threadIdx.x; n < N; n += 256) {
#pragma unroll
        for (int f = 0; f < 5; ++f) {
            float v = x[n * 5 + f];
            s[f] += v;
            q[f] += v * v;
        }
    }
    for (int f = 0; f < 10; ++f) {
        red[threadIdx.x] = (f < 5) ? s[f] : q[f - 5];
        __syncthreads();
        for (int off = 128; off > 0; off >>= 1) {
            if (threadIdx.x < off) red[threadIdx.x] += red[threadIdx.x + off];
            __syncthreads();
        }
        if (threadIdx.x == 0) stats[f] = red[0];
        __syncthreads();
    }
    if (threadIdx.x == 0) {
#pragma unroll
        for (int f = 0; f < 5; ++f) {
            float mu  = stats[f] / (float)N;
            float var = stats[5 + f] / (float)N - mu * mu;
            float sc  = g[f] * rsqrtf(var + 1e-5f);
            stats[16 + f] = sc;
            stats[24 + f] = b[f] - mu * sc;
        }
    }
}

// normalize features -> bf16 rows (stride 8, padded), and deg -> deg_inv in place
__global__ void k_bn_apply(const float* __restrict__ x, const float* __restrict__ stats,
                           unsigned short* __restrict__ x0, float* __restrict__ deg, int N) {
    int n = blockIdx.x * blockDim.x + threadIdx.x;
    if (n >= N) return;
#pragma unroll
    for (int f = 0; f < 5; ++f)
        x0[n * 8 + f] = f2bf(x[n * 5 + f] * stats[16 + f] + stats[24 + f]);
    x0[n * 8 + 5] = 0;
    x0[n * 8 + 6] = 0;
    x0[n * 8 + 7] = 0;
    float d = deg[n];
    deg[n] = 1.0f / (d > 1.0f ? d : 1.0f);
}

// Pre-shuffle a [K2F, C] f32 weight matrix into WMMA bf16 B fragments.
// 16-bit B layout: lane holds one column; lanes 0-15 -> K 0..7,16..23 of the 32-K
// tile, lanes 16-31 -> K 8..15,24..31. 16 bf16 per lane per fragment.
__global__ void k_prep_w(const float* __restrict__ W, int twoF, int C,
                         int NT, int KT, unsigned short* __restrict__ out, int total) {
    int i = blockIdx.x * blockDim.x + threadIdx.x;
    if (i >= total) return;
    int elem = i & 15;
    int lane = (i >> 4) & 31;
    int frag = i >> 9;
    int kt   = frag % KT;
    int nt   = frag / KT;
    int k    = kt * 32 + ((lane >= 16) ? 8 : 0) + (elem & 7) + ((elem >> 3) << 4);
    int n    = nt * 16 + (lane & 15);
    float v  = (k < twoF) ? W[k * C + n] : 0.0f;
    out[i]   = f2bf(v);
}

// y = acc * deg_inv -> bf16 node features; 2 channels per thread (packed cvt)
__global__ void k_scale(const float* __restrict__ acc, const float* __restrict__ dinv,
                        unsigned* __restrict__ xout, int totalPairs, int C) {
    int idx = blockIdx.x * blockDim.x + threadIdx.x;
    if (idx >= totalPairs) return;
    int i2   = idx * 2;
    float dv = dinv[i2 / C];
    xout[idx] = pack2bf(acc[i2] * dv, acc[i2 + 1] * dv);
}

// ---------------------------------------------------------------------------
// Fused EdgeConv: async gather -> WMMA MLP (2 layers, ReLU) -> atomic scatter
// One 16-edge tile per wave; 4 waves per block; grid divides E exactly so
// EXEC is all-ones at every WMMA.
// ---------------------------------------------------------------------------
template <int F_IN, int C1, int C2>
__global__ __launch_bounds__(128) void k_edgeconv(
    const unsigned short* __restrict__ xin, const int* __restrict__ src,
    const int* __restrict__ dst, const float* __restrict__ b1,
    const float* __restrict__ b2, const unsigned short* __restrict__ w1f,
    const unsigned short* __restrict__ w2f, float* __restrict__ yacc) {
    constexpr int WAVES = 4;
    constexpr int XSTR  = (F_IN < 8) ? 8 : F_IN;
    constexpr int K1P   = ((2 * F_IN + 31) / 32) * 32;
    constexpr int KT1 = K1P / 32, NT1 = C1 / 16;
    constexpr int KT2 = C1 / 32, NT2 = C2 / 16;

    __shared__ unsigned short ldsA[WAVES][16 * K1P];
    __shared__ unsigned short ldsH[WAVES][16 * C1];
    __shared__ int ldsD[WAVES][16];

    const int wave = threadIdx.x >> 5;
    const int lane = threadIdx.x & 31;
    const int e0   = (blockIdx.x * WAVES + wave) * 16;
    const int col  = lane & 15;
    const int hi   = lane >> 4;   // 0 or 1
    const int aoff = hi * 8;      // A-fragment K sub-offset

    __builtin_prefetch(w1f, 0, 3);
    __builtin_prefetch(w2f, 0, 3);

    // ---- phase 1: gather rows into A tile [16][K1P] bf16, row-major ----
    if constexpr (F_IN % 8 == 0) {
        // async global->LDS gather: xi into cols [0,F), xj into cols [F,2F)
        if (lane < 16) {
            int e = e0 + lane;
            int d = dst[e], s = src[e];
            ldsD[wave][lane] = d;
            unsigned short* row = &ldsA[wave][lane * K1P];
            const unsigned short* xi = xin + (size_t)d * XSTR;
            const unsigned short* xj = xin + (size_t)s * XSTR;
#pragma unroll
            for (int c = 0; c < F_IN / 8; ++c) {
                async_ld_b128(ldsoff(row + c * 8),        xi + c * 8);
                async_ld_b128(ldsoff(row + F_IN + c * 8), xj + c * 8);
            }
        }
        wait_async0();
        __syncthreads();
        // in-place: cols [F,2F) := xj - xi  (wave-local, packed dwords)
        {
            constexpr int HD = F_IN / 2;            // dwords per half-row
            unsigned* l32 = (unsigned*)&ldsA[wave][0];
#pragma unroll
            for (int t = lane; t < 16 * HD; t += 32) {
                int m = t / HD, c = t % HD;
                unsigned av = l32[m * (K1P / 2) + c];
                unsigned bv = l32[m * (K1P / 2) + HD + c];
                float dl = bf2f((unsigned short)(bv & 0xffff)) -
                           bf2f((unsigned short)(av & 0xffff));
                float dh = bf2f((unsigned short)(bv >> 16)) -
                           bf2f((unsigned short)(av >> 16));
                l32[m * (K1P / 2) + HD + c] = pack2bf(dl, dh);
            }
        }
    } else {
        // layer 0 (F=5): scalar path with zero padding to K1P
        if (lane < 16) {
            int e = e0 + lane;
            int d = dst[e], s = src[e];
            ldsD[wave][lane] = d;
            unsigned short* row = &ldsA[wave][lane * K1P];
            const unsigned short* xi = xin + (size_t)d * XSTR;
            const unsigned short* xj = xin + (size_t)s * XSTR;
            uv4 z = {0, 0, 0, 0};
#pragma unroll
            for (int c = 0; c < K1P / 8; ++c) *(uv4*)(row + c * 8) = z;
#pragma unroll
            for (int f = 0; f < F_IN; ++f) {
                unsigned short a = xi[f], b = xj[f];
                row[f]        = a;
                row[F_IN + f] = f2bf(bf2f(b) - bf2f(a));
            }
        }
    }
    __syncthreads();

    // ---- phase 2: h = relu(A @ W1 + b1) -> ldsH [16][C1] bf16 ----
#pragma unroll
    for (int nt = 0; nt < NT1; ++nt) {
        float bias = b1[nt * 16 + col];
        v8f acc;
#pragma unroll
        for (int i = 0; i < 8; ++i) acc[i] = bias;
#pragma unroll
        for (int kt = 0; kt < KT1; ++kt) {
            Frag A, B;
            const unsigned short* ap = &ldsA[wave][col * K1P + kt * 32 + aoff];
            A.q[0] = *(const uv4*)(ap);
            A.q[1] = *(const uv4*)(ap + 16);
            const unsigned short* bp = w1f + ((size_t)(nt * KT1 + kt) * 32 + lane) * 16;
            B.q[0] = *(const uv4*)(bp);
            B.q[1] = *(const uv4*)(bp + 8);
            acc = __builtin_amdgcn_wmma_f32_16x16x32_bf16(false, A.v, false, B.v,
                                                          (short)0, acc, false, false);
        }
#pragma unroll
        for (int r = 0; r < 8; r += 2) {
            unsigned u = pack2bf(fmaxf(acc[r], 0.0f), fmaxf(acc[r + 1], 0.0f));
            ldsH[wave][(r + hi * 8) * C1 + nt * 16 + col]     = (unsigned short)u;
            ldsH[wave][(r + 1 + hi * 8) * C1 + nt * 16 + col] = (unsigned short)(u >> 16);
        }
    }
    __syncthreads();

    // ---- phase 3: y = relu(h @ W2 + b2), atomic scatter-add to dst nodes ----
#pragma unroll
    for (int nt = 0; nt < NT2; ++nt) {
        float bias = b2[nt * 16 + col];
        v8f acc;
#pragma unroll
        for (int i = 0; i < 8; ++i) acc[i] = bias;
#pragma unroll
        for (int kt = 0; kt < KT2; ++kt) {
            Frag A, B;
            const unsigned short* ap = &ldsH[wave][col * C1 + kt * 32 + aoff];
            A.q[0] = *(const uv4*)(ap);
            A.q[1] = *(const uv4*)(ap + 16);
            const unsigned short* bp = w2f + ((size_t)(nt * KT2 + kt) * 32 + lane) * 16;
            B.q[0] = *(const uv4*)(bp);
            B.q[1] = *(const uv4*)(bp + 8);
            acc = __builtin_amdgcn_wmma_f32_16x16x32_bf16(false, A.v, false, B.v,
                                                          (short)0, acc, false, false);
        }
#pragma unroll
        for (int r = 0; r < 8; ++r) {
            float v = fmaxf(acc[r], 0.0f);
            int d   = ldsD[wave][r + hi * 8];
            atomicAdd(&yacc[(size_t)d * C2 + nt * 16 + col], v);
        }
    }
}

// ---------------------------------------------------------------------------
// Readout: centroid distances + per-graph pooling + FC
// ---------------------------------------------------------------------------
__global__ void k_cnt(const int* __restrict__ gid, float* __restrict__ cnt, int N) {
    int n = blockIdx.x * blockDim.x + threadIdx.x;
    if (n < N) atomicAdd(&cnt[gid[n]], 1.0f);
}

__global__ void k_dist(const float* __restrict__ acc, const float* __restrict__ dinv,
                       const float* __restrict__ cent, const int* __restrict__ gid,
                       float* __restrict__ pooled, int N) {
    int i = blockIdx.x * blockDim.x + threadIdx.x;
    if (i >= N * N_CENT) return;
    int n = i / N_CENT, c = i % N_CENT;
    float sc = dinv[n];
    float s  = 0.0f;
#pragma unroll 8
    for (int d = 0; d < D_FIN; ++d) {
        float t = acc[(size_t)n * D_FIN + d] * sc - cent[c * D_FIN + d];
        s += t * t;
    }
    atomicAdd(&pooled[gid[n] * N_CENT + c], sqrtf(s + 1e-12f));
}

__global__ void k_fc(const float* __restrict__ pooled, const float* __restrict__ cnt,
                     const float* __restrict__ w, const float* __restrict__ b,
                     float* __restrict__ out) {
    int i = blockIdx.x * blockDim.x + threadIdx.x;
    if (i >= N_GRAPHS * 2) return;
    int g = i >> 1, k = i & 1;
    float cv  = cnt[g];
    float inv = 1.0f / (cv > 1.0f ? cv : 1.0f);
    float s   = b[k];
    for (int c = 0; c < N_CENT; ++c) s += pooled[g * N_CENT + c] * inv * w[c * 2 + k];
    out[i] = s;
}

// ---------------------------------------------------------------------------
// Launch
// ---------------------------------------------------------------------------
extern "C" void kernel_launch(void* const* d_in, const int* in_sizes, int n_in,
                              void* d_out, int out_size, void* d_ws, size_t ws_size,
                              hipStream_t stream) {
    (void)in_sizes; (void)n_in; (void)out_size; (void)ws_size;
    const float* feat = (const float*)d_in[0];
    const int*   src  = (const int*)d_in[1];
    const int*   dst  = (const int*)d_in[2];
    const int*   gid  = (const int*)d_in[3];
    const float* bng  = (const float*)d_in[4];
    const float* bnb  = (const float*)d_in[5];
    const float* cent = (const float*)d_in[30];
    const float* fcw  = (const float*)d_in[31];
    const float* fcb  = (const float*)d_in[32];
    float*       out  = (float*)d_out;

    char* ws = (char*)d_ws;
    unsigned short* xba   = (unsigned short*)(ws + OFF_XBA);
    unsigned short* xbb   = (unsigned short*)(ws + OFF_XBB);
    float*          acc   = (float*)(ws + OFF_ACC);
    float*          deg   = (float*)(ws + OFF_DEG);
    float*          stats = (float*)(ws + OFF_STATS);
    float*          pool  = (float*)(ws + OFF_POOL);
    float*          cnt   = (float*)(ws + OFF_CNT);

    const int N = N_NODES, E = N_EDGES;

    // degree + batch norm
    k_zero_f32<<<(N + 255) / 256, 256, 0, stream>>>(deg, N);
    k_deg<<<(E + 255) / 256, 256, 0, stream>>>(dst, deg, E);
    k_bn_stats<<<1, 256, 0, stream>>>(feat, bng, bnb, stats, N);
    k_bn_apply<<<(N + 255) / 256, 256, 0, stream>>>(feat, stats, xba, deg, N);

    // weight fragment prep
    const int Fs[6]  = {5, 32, 32, 64, 64, 128};
    const int C1s[6] = {32, 32, 64, 64, 128, 128};
    const int C2s[6] = {32, 32, 64, 64, 128, 128};
    unsigned short* w1f[6];
    unsigned short* w2f[6];
    for (int l = 0; l < 6; ++l) {
        const float* W1 = (const float*)d_in[6 + 4 * l];
        const float* W2 = (const float*)d_in[8 + 4 * l];
        int F = Fs[l], C1 = C1s[l], C2 = C2s[l];
        int K1P = ((2 * F + 31) / 32) * 32;
        int NT1 = C1 / 16, KT1 = K1P / 32, NT2 = C2 / 16, KT2 = C1 / 32;
        w1f[l] = (unsigned short*)(ws + OFF_WF + (size_t)l * 131072);
        w2f[l] = w1f[l] + 65536 / 2;
        int t1 = NT1 * KT1 * 512, t2 = NT2 * KT2 * 512;
        k_prep_w<<<(t1 + 255) / 256, 256, 0, stream>>>(W1, 2 * F, C1, NT1, KT1, w1f[l], t1);
        k_prep_w<<<(t2 + 255) / 256, 256, 0, stream>>>(W2, C1, C2, NT2, KT2, w2f[l], t2);
    }

    const int convBlocks = E / 16 / 4;  // 7500, exact
    unsigned short* xc = xba;
    unsigned short* xn = xbb;

    for (int l = 0; l < 6; ++l) {
        const float* b1 = (const float*)d_in[7 + 4 * l];
        const float* b2 = (const float*)d_in[9 + 4 * l];
        int C2 = C2s[l];
        k_zero_f32<<<(N * C2 + 255) / 256, 256, 0, stream>>>(acc, N * C2);
        switch (l) {
            case 0: k_edgeconv<5, 32, 32><<<convBlocks, 128, 0, stream>>>(xc, src, dst, b1, b2, w1f[0], w2f[0], acc); break;
            case 1: k_edgeconv<32, 32, 32><<<convBlocks, 128, 0, stream>>>(xc, src, dst, b1, b2, w1f[1], w2f[1], acc); break;
            case 2: k_edgeconv<32, 64, 64><<<convBlocks, 128, 0, stream>>>(xc, src, dst, b1, b2, w1f[2], w2f[2], acc); break;
            case 3: k_edgeconv<64, 64, 64><<<convBlocks, 128, 0, stream>>>(xc, src, dst, b1, b2, w1f[3], w2f[3], acc); break;
            case 4: k_edgeconv<64, 128, 128><<<convBlocks, 128, 0, stream>>>(xc, src, dst, b1, b2, w1f[4], w2f[4], acc); break;
            case 5: k_edgeconv<128, 128, 128><<<convBlocks, 128, 0, stream>>>(xc, src, dst, b1, b2, w1f[5], w2f[5], acc); break;
        }
        if (l < 5) {
            k_scale<<<(N * C2 / 2 + 255) / 256, 256, 0, stream>>>(acc, deg, (unsigned*)xn, N * C2 / 2, C2);
            unsigned short* t = xc; xc = xn; xn = t;
        }
    }

    // readout: centroid distances, per-graph mean pooling, FC
    k_zero_f32<<<(N_GRAPHS * N_CENT + 255) / 256, 256, 0, stream>>>(pool, N_GRAPHS * N_CENT);
    k_zero_f32<<<1, 256, 0, stream>>>(cnt, N_GRAPHS);
    k_cnt<<<(N + 255) / 256, 256, 0, stream>>>(gid, cnt, N);
    k_dist<<<(N * N_CENT + 255) / 256, 256, 0, stream>>>(acc, deg, cent, gid, pool, N);
    k_fc<<<1, 256, 0, stream>>>(pool, cnt, fcw, fcb, out);
}
